// FNOBlockNd_34325378630286
// MI455X (gfx1250) — matrix operand order repaired and checked by reading, not compiled, not run on hardware
//
#include <hip/hip_runtime.h>

// ---------------------------------------------------------------------------
// FNO block for MI455X (gfx1250, wave32, WMMA bf16 16x16x32, f32 accumulate)
// ---------------------------------------------------------------------------

typedef __attribute__((ext_vector_type(16))) __bf16 v16bf;
typedef __attribute__((ext_vector_type(8)))  float  v8f;

using uint   = unsigned int;
using ushort = unsigned short;

union BF16x16 { v16bf v; ushort u[16]; uint d[8]; };

#define DEVINL __device__ __forceinline__

static constexpr int   BATCH = 16, CIN = 64, COUT = 64, HH = 128, WW = 128;
static constexpr float TWO_PI = 6.28318530717958647692f;

// ---- workspace layout (bytes) ----
static constexpr size_t OFF_BASIS = 0;                       // ushort[32][128] fwd cos/sin basis
static constexpr size_t OFF_INVA  = 8192;                    // ushort[128][32] inverse stage-1 A basis
static constexpr size_t OFF_INVB  = 16384;                   // ushort[128][32] inverse stage-2 B basis (alpha/128 folded)
static constexpr size_t OFF_PROJ  = 24576;                   // float[16][64]   temb projection
static constexpr size_t OFF_FRE   = 32768;                   // float[16][16][16][64]  x_fft real  [k][l][b][i]
static constexpr size_t OFF_FIM   = OFF_FRE + 1048576;       // float  x_fft imag
static constexpr size_t OFF_GRE   = OFF_FIM + 1048576;       // float[16][64][16][16]  out_fft real [b][o][k][l]
static constexpr size_t OFF_GIM   = OFF_GRE + 1048576;       // float  out_fft imag
static constexpr size_t OFF_WTR   = OFF_GIM + 1048576;       // ushort[256][64][64] weight real [mode][o][i]
static constexpr size_t OFF_WTI   = OFF_WTR + 2097152;       // ushort weight imag
// total ~8.4 MB

DEVINL ushort f2bf(float f) {
  union { float f; uint u; } c; c.f = f;
  uint u = c.u;
  uint r = (u + 0x7FFFu + ((u >> 16) & 1u)) >> 16;   // round-to-nearest-even
  return (ushort)r;
}

DEVINL v8f vzero() { v8f z = {0.f,0.f,0.f,0.f,0.f,0.f,0.f,0.f}; return z; }

DEVINL v8f wmma_bf16(v16bf a, v16bf b, v8f c) {
  return __builtin_amdgcn_wmma_f32_16x16x32_bf16(false, a, false, b, (short)0, c, false, false);
}

// Async global->LDS copy of 16 bytes per lane (CDNA5 ASYNCcnt path).
// LDS destination address = low 32 bits of the generic shared pointer
// global source = 64-bit VGPR pair.
DEVINL void async_copy_b128(ushort* lds_dst, const ushort* g_src) {
  uint ldsa = (uint)(unsigned long long)lds_dst;
  asm volatile("global_load_async_to_lds_b128 %0, %1, off"
               :: "v"(ldsa), "v"(g_src) : "memory");
}

DEVINL void async_wait0() {
  asm volatile("s_wait_asynccnt 0" ::: "memory");
}

// A fragment (16x32 bf16), source row-major [M][K], K contiguous.
// lane L: m = L&15, k-half = L>>4; VGPR r: K = (L>>4)*16 + 2r
DEVINL v16bf lds_load_A(const ushort* base, int stride, int m0, int k0) {
  const int lane = threadIdx.x & 31;
  const ushort* row = base + (size_t)(m0 + (lane & 15)) * stride;
  const int kb = k0 + (lane >> 4) * 16;
  BF16x16 t;
#pragma unroll
  for (int r = 0; r < 8; ++r) {
    int k = kb + 2 * r;
    t.d[r] = *(const uint*)(row + k);
  }
  return t.v;
}

// B fragment (32x16 bf16), source layout [N][K], K contiguous.
// lane L: n = L&15, VGPR r: K = (L>>4)*16 + 2r
DEVINL v16bf lds_load_B(const ushort* base, int stride, int n0, int k0) {
  const int lane = threadIdx.x & 31;
  const ushort* row = base + (size_t)(n0 + (lane & 15)) * stride;
  const int kb = k0 + (lane >> 4) * 16;
  BF16x16 t;
#pragma unroll
  for (int r = 0; r < 8; ++r) t.d[r] = *(const uint*)(row + kb + 2 * r);
  return t.v;
}

// D fragment (16x16 f32): lane<16 -> N=lane, M=r ; lane>=16 -> N=lane-16, M=8+r
DEVINL void store_D_f32(float* out, int stride, int m0, int n0, v8f c) {
  const int lane = threadIdx.x & 31;
  const int n = n0 + (lane & 15);
  const int m = m0 + (lane >> 4) * 8;
#pragma unroll
  for (int r = 0; r < 8; ++r) out[(size_t)(m + r) * stride + n] = c[r];
}

DEVINL void store_D_bf16(ushort* out, int stride, int m0, int n0, v8f c) {
  const int lane = threadIdx.x & 31;
  const int n = n0 + (lane & 15);
  const int m = m0 + (lane >> 4) * 8;
#pragma unroll
  for (int r = 0; r < 8; ++r) out[(size_t)(m + r) * stride + n] = f2bf(c[r]);
}

// transposed store: element (m,n) -> out[n][m]
DEVINL void store_D_bf16_T(ushort* out, int stride, int m0, int n0, v8f c) {
  const int lane = threadIdx.x & 31;
  const int n = n0 + (lane & 15);
  const int m = m0 + (lane >> 4) * 8;
#pragma unroll
  for (int r = 0; r < 8; ++r) out[(size_t)n * stride + (m + r)] = f2bf(c[r]);
}

// ---------------------------------------------------------------------------
// Kernel 0: DFT basis tables (bf16) + temb projection  proj = silu(t) @ Wt^T + b
// grid: 17 blocks x 256. blocks 0..15 -> proj rows, block 16 -> tables.
// ---------------------------------------------------------------------------
__global__ void k_setup(const float* __restrict__ t_emb,
                        const float* __restrict__ temb_w,
                        const float* __restrict__ temb_b,
                        char* __restrict__ ws) {
  const int tid = threadIdx.x;
  if (blockIdx.x == 16) {
    ushort* basis = (ushort*)(ws + OFF_BASIS);
    ushort* invA  = (ushort*)(ws + OFF_INVA);
    ushort* invB  = (ushort*)(ws + OFF_INVB);
    const float inv128 = 1.0f / 128.0f;
#pragma unroll 1
    for (int j = 0; j < 16; ++j) {
      int e = tid + (j << 8);
      { // basis[k][h]: rows 0..15 cos, 16..31 sin of 2*pi*l*h/128
        int k = e >> 7, h = e & 127, l = k & 15;
        float ang = TWO_PI * (float)(l * h) * inv128;
        basis[e] = f2bf((k < 16) ? __cosf(ang) : __sinf(ang));
      }
      { // invA[h][kk]
        int h = e >> 5, kk = e & 31, l = kk & 15;
        float ang = TWO_PI * (float)(l * h) * inv128;
        invA[e] = f2bf((kk < 16) ? __cosf(ang) : __sinf(ang));
      }
      { // invB[w][kk], alpha_l/128 folded, sin part negated
        int w = e >> 5, kk = e & 31, l = kk & 15;
        float al = ((l == 0) ? 1.0f : 2.0f) * 0.0078125f;
        float ang = TWO_PI * (float)(l * w) * inv128;
        invB[e] = f2bf((kk < 16) ? al * __cosf(ang) : -al * __sinf(ang));
      }
    }
  } else {
    float* proj = (float*)(ws + OFF_PROJ);
    const int b = blockIdx.x;
    if (tid < COUT) {
      float acc = temb_b[tid];
#pragma unroll 4
      for (int s = 0; s < 512; ++s) {
        float tv = t_emb[b * 512 + s];
        float sl = tv / (1.0f + __expf(-tv));      // silu
        acc += sl * temb_w[tid * 512 + s];
      }
      proj[b * COUT + tid] = acc;
    }
  }
}

// ---------------------------------------------------------------------------
// Kernel 1: weight transpose  [i][o][k][l] f32 -> [mode][o][i] bf16
// grid: 256 blocks (one per mode) x 256
// ---------------------------------------------------------------------------
__global__ void k_wt(const float* __restrict__ wr, const float* __restrict__ wi,
                     char* __restrict__ ws) {
  ushort* Wtr = (ushort*)(ws + OFF_WTR);
  ushort* Wti = (ushort*)(ws + OFF_WTI);
  const int m = blockIdx.x;
  const int tid = threadIdx.x;
#pragma unroll
  for (int j = 0; j < 16; ++j) {
    int e = tid + (j << 8);               // e = o*64 + i
    int o = e >> 6, i = e & 63;
    size_t src = (size_t)(i * 64 + o) * 256 + m;
    Wtr[(size_t)m * 4096 + e] = f2bf(wr[src]);
    Wti[(size_t)m * 4096 + e] = f2bf(wi[src]);
  }
}

// ---------------------------------------------------------------------------
// Kernel 2: forward truncated DFT.  grid: 1024 blocks (b,ci) x 256 (8 waves)
// stage1: T[h][l] = sum_w x[h][w]*basis[l][w]  (l<16 cos, l>=16 sin)
// stage2: Fre = (Ch@U - Sh@V)/128,  Fim = -(Ch@V + Sh@U)/128
// ---------------------------------------------------------------------------
__global__ void k_fwd(const float* __restrict__ x, char* __restrict__ ws) {
  __shared__ alignas(16) ushort xl[128 * 128];   // x tile bf16, [h][w]
  __shared__ alignas(16) ushort bl[32 * 128];    // basis copy
  __shared__ alignas(16) ushort Tt[32 * 128];    // stage1 result, transposed [l][h]
  __shared__ alignas(16) float  Pm[4 * 256];     // stage2 partial products

  const ushort* basisg = (const ushort*)(ws + OFF_BASIS);
  float* FreG = (float*)(ws + OFF_FRE);
  float* FimG = (float*)(ws + OFF_FIM);

  const int b  = blockIdx.x >> 6;
  const int ci = blockIdx.x & 63;
  const int tid = threadIdx.x;
  const float* xp = x + ((size_t)(b * CIN + ci) << 14);

  // basis: pure copy -> async DMA to LDS (ASYNCcnt path), 16 B per lane
#pragma unroll
  for (int j = 0; j < 2; ++j) {
    int e8 = (tid + (j << 8)) * 8;                 // ushort offset, 16B chunks
    async_copy_b128(bl + e8, basisg + e8);
  }
  // x tile: f32 -> bf16 conversion overlaps the async copies
#pragma unroll 8
  for (int j = 0; j < 64; ++j) { int e = tid + (j << 8); xl[e] = f2bf(xp[e]); }
  async_wait0();
  __syncthreads();

  const int wave = tid >> 5;
  { // stage 1: wave = M-tile over h
    const int mt = wave;
#pragma unroll
    for (int nt = 0; nt < 2; ++nt) {
      v8f acc = vzero();
#pragma unroll
      for (int ks = 0; ks < 4; ++ks)
        acc = wmma_bf16(lds_load_A(xl, 128, mt * 16, ks * 32),
                        lds_load_B(bl, 128, nt * 16, ks * 32), acc);
      store_D_bf16_T(Tt, 128, mt * 16, nt * 16, acc);
    }
  }
  __syncthreads();

  if (wave < 4) { // stage 2: p0=Ch@U p1=Sh@V p2=Ch@V p3=Sh@U
    const int p  = wave;
    const int ar = (p == 0 || p == 2) ? 0 : 16;
    const int br = (p == 0 || p == 3) ? 0 : 16;
    v8f acc = vzero();
#pragma unroll
    for (int ks = 0; ks < 4; ++ks)
      acc = wmma_bf16(lds_load_A(bl, 128, ar, ks * 32),
                      lds_load_B(Tt, 128, br, ks * 32), acc);
    store_D_f32(Pm + p * 256, 16, 0, 0, acc);
  }
  __syncthreads();

  { // combine & scatter to [k][l][b][i]
    float re = (Pm[tid] - Pm[256 + tid]) * 0.0078125f;
    float im = -(Pm[512 + tid] + Pm[768 + tid]) * 0.0078125f;
    size_t addr = (size_t)tid * (BATCH * CIN) + b * CIN + ci;
    FreG[addr] = re;
    FimG[addr] = im;
  }
}

// ---------------------------------------------------------------------------
// Kernel 3: per-mode complex channel mixing. grid: 256 blocks (modes) x 256
// Gre = Xr@Wr - Xi@Wi ;  Gim = Xr@Wi + Xi@Wr        ([16b x 64i] @ [64i x 64o])
// ---------------------------------------------------------------------------
__global__ void k_mix(char* __restrict__ ws) {
  __shared__ alignas(16) ushort Xr[16 * 64], Xi[16 * 64];
  __shared__ alignas(16) ushort Wrl[64 * 64], Wil[64 * 64];
  __shared__ alignas(16) float  Pm[4][16 * 64];

  const float*  FreG = (const float*)(ws + OFF_FRE);
  const float*  FimG = (const float*)(ws + OFF_FIM);
  const ushort* WtrG = (const ushort*)(ws + OFF_WTR);
  const ushort* WtiG = (const ushort*)(ws + OFF_WTI);
  float* GreG = (float*)(ws + OFF_GRE);
  float* GimG = (float*)(ws + OFF_GIM);

  const int m = blockIdx.x;
  const int tid = threadIdx.x;

  // weight tiles: pure bf16 copies -> async DMA to LDS
#pragma unroll
  for (int j = 0; j < 2; ++j) {
    int e8 = (tid + (j << 8)) * 8;
    async_copy_b128(Wrl + e8, WtrG + (size_t)m * 4096 + e8);
    async_copy_b128(Wil + e8, WtiG + (size_t)m * 4096 + e8);
  }
  // spectral activations: f32 -> bf16 conversion
#pragma unroll
  for (int j = 0; j < 4; ++j) {
    int e = tid + (j << 8);
    Xr[e] = f2bf(FreG[(size_t)m * 1024 + e]);
    Xi[e] = f2bf(FimG[(size_t)m * 1024 + e]);
  }
  async_wait0();
  __syncthreads();

  const int wave = tid >> 5;
  const int p  = wave >> 1;
  const int nh = wave & 1;
  const ushort* as = (p == 0 || p == 2) ? Xr : Xi;   // p0:XrWr p1:XiWi p2:XrWi p3:XiWr
  const ushort* bs = (p == 0 || p == 3) ? Wrl : Wil;
#pragma unroll
  for (int t = 0; t < 2; ++t) {
    int nt = nh * 2 + t;
    v8f acc = vzero();
#pragma unroll
    for (int ks = 0; ks < 2; ++ks)
      acc = wmma_bf16(lds_load_A(as, 64, 0, ks * 32),
                      lds_load_B(bs, 64, nt * 16, ks * 32), acc);
    store_D_f32(&Pm[p][0], 64, 0, nt * 16, acc);
  }
  __syncthreads();

#pragma unroll
  for (int j = 0; j < 4; ++j) {
    int e = tid * 4 + j;                      // e = b*64 + o
    float gre = Pm[0][e] - Pm[1][e];
    float gim = Pm[2][e] + Pm[3][e];
    size_t addr = (size_t)e * 256 + m;        // [b][o][k][l]
    GreG[addr] = gre;
    GimG[addr] = gim;
  }
}

// ---------------------------------------------------------------------------
// Kernel 4: inverse truncated DFT -> spectral field written into d_out.
// grid: 1024 blocks (b,o) x 256.
// stage1: P = cos@Gre - sin@Gim, Q = sin@Gre + cos@Gim (K stacked to 32)
// stage2: out[h][w] = sum_l alpha_l*(P cos - Q sin)/128  (folded into invB)
// ---------------------------------------------------------------------------
__global__ void k_inv(char* __restrict__ ws, float* __restrict__ outg) {
  __shared__ alignas(16) ushort BP[16 * 32], BQ[16 * 32];  // [l][kk]
  __shared__ alignas(16) ushort S2A[128 * 32];             // [h][kk]: kk<16 P, kk>=16 Q
  __shared__ alignas(16) ushort iA[128 * 32], iB[128 * 32];

  const float*  GreG = (const float*)(ws + OFF_GRE);
  const float*  GimG = (const float*)(ws + OFF_GIM);
  const ushort* invAg = (const ushort*)(ws + OFF_INVA);
  const ushort* invBg = (const ushort*)(ws + OFF_INVB);

  const int b = blockIdx.x >> 6;
  const int o = blockIdx.x & 63;
  const int tid = threadIdx.x;

  // basis tables: pure copies -> async DMA to LDS
#pragma unroll
  for (int j = 0; j < 2; ++j) {
    int e8 = (tid + (j << 8)) * 8;
    async_copy_b128(iA + e8, invAg + e8);
    async_copy_b128(iB + e8, invBg + e8);
  }
  {
    size_t base = (size_t)(b * COUT + o) * 256;
    int k = tid >> 4, l = tid & 15;
    float gr = GreG[base + tid];
    float gi = GimG[base + tid];
    BP[l * 32 + k]      = f2bf(gr);
    BP[l * 32 + 16 + k] = f2bf(-gi);
    BQ[l * 32 + k]      = f2bf(gi);
    BQ[l * 32 + 16 + k] = f2bf(gr);
  }
  async_wait0();
  __syncthreads();

  const int mt = tid >> 5;   // wave = h-tile
  {
    v16bf a1 = lds_load_A(iA, 32, mt * 16, 0);
    v8f accP = wmma_bf16(a1, lds_load_B(BP, 32, 0, 0), vzero());
    store_D_bf16(S2A, 32, mt * 16, 0, accP);
    v8f accQ = wmma_bf16(a1, lds_load_B(BQ, 32, 0, 0), vzero());
    store_D_bf16(S2A, 32, mt * 16, 16, accQ);
  }
  __syncthreads();

  float* op = outg + ((size_t)(b * COUT + o) << 14);
  v16bf a2 = lds_load_A(S2A, 32, mt * 16, 0);
#pragma unroll
  for (int wt = 0; wt < 8; ++wt) {
    v8f acc = wmma_bf16(a2, lds_load_B(iB, 32, wt * 16, 0), vzero());
    store_D_f32(op, 128, mt * 16, wt * 16, acc);
  }
}

// ---------------------------------------------------------------------------
// Kernel 5: 1x1-conv bypass (WMMA over channels) fused with spectral add,
// exact GELU, bias and temb projection. grid: 1024 blocks (b, 2-row h tile).
// x tile stored TRANSPOSED in LDS ([hw][i], padded stride 66) so B fragments
// are contiguous-K -> 2x ds_load_b128 instead of 16x ds_load_u16.
// out[o][hw] = gelu(spec + sum_i bw[o][i]*x[i][hw] + bb[o]) + proj[b][o]
// ---------------------------------------------------------------------------
__global__ void k_byp(const float* __restrict__ x,
                      const float* __restrict__ bw,
                      const float* __restrict__ bbias,
                      char* __restrict__ ws,
                      float* __restrict__ outg) {
  static constexpr int XT_STRIDE = 66;                 // 64 + 2 pad (bank decorrelation)
  __shared__ alignas(16) ushort xlT[256 * XT_STRIDE];  // [hw][i]
  __shared__ alignas(16) ushort bwl[64 * 64];          // [o][i]

  const float* proj = (const float*)(ws + OFF_PROJ);

  const int b  = blockIdx.x >> 6;
  const int ht = blockIdx.x & 63;
  const int h0 = ht * 2;
  const int tid = threadIdx.x;

#pragma unroll
  for (int j = 0; j < 16; ++j) { int e = tid + (j << 8); bwl[e] = f2bf(bw[e]); }
#pragma unroll 8
  for (int j = 0; j < 64; ++j) {
    int e = tid + (j << 8);
    int i = e >> 8, n = e & 255;                       // coalesced global read
    xlT[n * XT_STRIDE + i] =                            // transposing LDS write
        f2bf(x[((size_t)(b * CIN + i) << 14) + h0 * 128 + n]);
  }
  __syncthreads();

  const int wave = tid >> 5;
  const int lane = tid & 31;
  const int lj = lane & 15;
  const int lh = lane >> 4;

#pragma unroll
  for (int t = 0; t < 2; ++t) {
    const int nt = wave * 2 + t;
    const int col = nt * 16 + lj;
    const int hloc = col >> 7;
    const int wcol = col & 127;
    v16bf b0 = lds_load_B(xlT, XT_STRIDE, nt * 16, 0);
    v16bf b1 = lds_load_B(xlT, XT_STRIDE, nt * 16, 32);
#pragma unroll
    for (int mt = 0; mt < 4; ++mt) {
      v8f acc = wmma_bf16(lds_load_A(bwl, 64, mt * 16, 0),  b0, vzero());
      acc     = wmma_bf16(lds_load_A(bwl, 64, mt * 16, 32), b1, acc);
#pragma unroll
      for (int r = 0; r < 8; ++r) {
        int o = mt * 16 + lh * 8 + r;
        size_t idx = (((size_t)(b * COUT + o)) << 14) + (size_t)(h0 + hloc) * 128 + wcol;
        float v = outg[idx] + acc[r] + bbias[o];
        float g = 0.5f * v * (1.0f + erff(v * 0.70710678118654752f));
        outg[idx] = g + proj[b * COUT + o];
      }
    }
  }
}

// ---------------------------------------------------------------------------
extern "C" void kernel_launch(void* const* d_in, const int* in_sizes, int n_in,
                              void* d_out, int out_size, void* d_ws, size_t ws_size,
                              hipStream_t stream) {
  (void)in_sizes; (void)n_in; (void)out_size; (void)ws_size;
  const float* x      = (const float*)d_in[0];
  const float* t_emb  = (const float*)d_in[1];
  const float* w_real = (const float*)d_in[2];
  const float* w_imag = (const float*)d_in[3];
  const float* byp_w  = (const float*)d_in[4];
  const float* byp_b  = (const float*)d_in[5];
  const float* temb_w = (const float*)d_in[6];
  const float* temb_b = (const float*)d_in[7];
  float* out = (float*)d_out;
  char*  ws  = (char*)d_ws;

  k_setup<<<17,   256, 0, stream>>>(t_emb, temb_w, temb_b, ws);
  k_wt   <<<256,  256, 0, stream>>>(w_real, w_imag, ws);
  k_fwd  <<<1024, 256, 0, stream>>>(x, ws);
  k_mix  <<<256,  256, 0, stream>>>(ws);
  k_inv  <<<1024, 256, 0, stream>>>(ws, out);
  k_byp  <<<1024, 256, 0, stream>>>(x, byp_w, byp_b, ws, out);
}
